// DeStripeModel_torch_42460046688472
// MI455X (gfx1250) — matrix operand order repaired and checked by reading, not compiled, not run on hardware
//
#include <hip/hip_runtime.h>
#include <cmath>

// ---------------- problem constants ----------------
#define M_L   1025
#define N_L   1025
#define HW2   525312          // M_L*N_L/2
#define NPIX  1050625         // M_L*N_L
#define INC   16
#define N_G   131072
#define N_MASK (HW2 - N_G)
#define MH    3072
#define NH    3072
#define PI_F  3.14159265358979323846f

__device__ __constant__ float kAngles[3] = {-10.0f, 0.0f, 10.0f};

// ---------------- small complex helpers ----------------
__device__ __forceinline__ float2 cmul(float2 a, float2 b) {
  return make_float2(a.x * b.x - a.y * b.y, a.x * b.y + a.y * b.x);
}
__device__ __forceinline__ float2 cadd(float2 a, float2 b) {
  return make_float2(a.x + b.x, a.y + b.y);
}
__device__ __forceinline__ float2 conj2(float2 a) { return make_float2(a.x, -a.y); }
__device__ __forceinline__ float2 crelu2(float2 a) {
  return make_float2(fmaxf(a.x, 0.f), fmaxf(a.y, 0.f));
}
__device__ __forceinline__ float2 cis(float t) {
  float s, c; __sincosf(t, &s, &c); return make_float2(c, s);
}

// ---------------- WMMA plumbing (gfx1250, wave32) ----------------
typedef __attribute__((ext_vector_type(16))) _Float16 v16h;
typedef __attribute__((ext_vector_type(8)))  float    v8f;
typedef __attribute__((ext_vector_type(2)))  float    v2f;

__device__ __forceinline__ v8f vzero8() {
  v8f z;
#pragma unroll
  for (int j = 0; j < 8; ++j) z[j] = 0.f;
  return z;
}

// --- f16 fragment geometry (V_WMMA_F32_16X16X32_F16) ---
// A frag: lane l holds row m=l&15; half h -> packed K index kk:
//   l<16 : kk = h (h<8) | h+8 (h>=8)      -> kk in {0..7,16..23}
//   l>=16: kk = h+8 (h<8) | h+16 (h>=8)   -> kk in {8..15,24..31}
// Inverse map (which lane/half owns packed index kk for row m):
__device__ __forceinline__ void frag_pos(int kk, int m, int& lane_a, int& h) {
  if (kk < 8)       { lane_a = m;      h = kk;      }
  else if (kk < 16) { lane_a = m + 16; h = kk - 8;  }
  else if (kk < 24) { lane_a = m;      h = kk - 8;  }
  else              { lane_a = m + 16; h = kk - 16; }
}

// Store complex value (row m, complex channel c) into a fragment-layout LDS
// buffer holding `ldc` complex channels packed along K as [re(0..ldc-1) | im].
// Buffer layout: [chunk][lane(32)][half(16)], chunk stride 512 halves.
__device__ __forceinline__ void frag_store(_Float16* base, int ldc, int m, int c, float2 v) {
  int la, h;
  int kre = c, kim = ldc + c;
  frag_pos(kre & 31, m, la, h);
  base[(kre >> 5) * 512 + la * 16 + h] = (_Float16)v.x;
  frag_pos(kim & 31, m, la, h);
  base[(kim >> 5) * 512 + la * 16 + h] = (_Float16)v.y;
}
__device__ __forceinline__ float2 frag_read(const _Float16* base, int ldc, int m, int c) {
  int la, h; float2 r;
  int kre = c, kim = ldc + c;
  frag_pos(kre & 31, m, la, h);
  r.x = (float)base[(kre >> 5) * 512 + la * 16 + h];
  frag_pos(kim & 31, m, la, h);
  r.y = (float)base[(kim >> 5) * 512 + la * 16 + h];
  return r;
}

// Pre-pack a complex weight (ldk x 16, row-major [k][n]) into B-fragment images.
// dst layout: [chunk][io(2)][lane(32)][half(16)]; io=0 real-out ([Wr;-Wi]),
// io=1 imag-out ([Wi;Wr]).  B frag: lane l -> n=l&15, half h -> K = c*32+((h&7)|((h&8)<<1))+(l<16?0:8).
__device__ void pack_b(const float2* __restrict__ Wg, int ldk, _Float16* dst,
                       int tid, int nthreads) {
  int nchunk = (2 * ldk) / 32;
  int total = nchunk * 1024;
  for (int i = tid; i < total; i += nthreads) {
    int chunk = i >> 10;
    int rem = i & 1023;
    int io = rem >> 9;
    int l = (rem & 511) >> 4;
    int h = rem & 15;
    int k = chunk * 32 + ((h & 7) | ((h & 8) << 1)) + ((l < 16) ? 0 : 8);
    float v;
    if (k < ldk) { float2 wv = Wg[k * 16 + (l & 15)]; v = io ? wv.y : wv.x; }
    else         { float2 wv = Wg[(k - ldk) * 16 + (l & 15)]; v = io ? wv.x : -wv.y; }
    dst[i] = (_Float16)v;
  }
}

// Complex GEMM from fragment-layout LDS buffers: one aligned 32B load per operand.
__device__ __forceinline__ void cgemm_f(const _Float16* fa, const _Float16* fb, int nchunk,
                                        int lane, v8f& dr, v8f& di) {
  dr = vzero8(); di = vzero8();
  for (int c = 0; c < nchunk; ++c) {
    v16h a  = *(const v16h*)(fa + c * 512 + lane * 16);
    v16h br = *(const v16h*)(fb + c * 1024 + lane * 16);
    dr = __builtin_amdgcn_wmma_f32_16x16x32_f16(false, a, false, br, (short)0, dr, false, false);
    v16h bi = *(const v16h*)(fb + c * 1024 + 512 + lane * 16);
    di = __builtin_amdgcn_wmma_f32_16x16x32_f16(false, a, false, bi, (short)0, di, false, false);
  }
}
// C/D layout: lane l -> n = l&15, VGPR j -> m = j + (l<16?0:8)

// ---------------- analytic TV-FFT constants ----------------
__device__ __forceinline__ float2 g_at(int r, int q, int isY) {
  int idx = isY ? r : q;
  int sh = (idx + 513) % 1025;             // undo fftshift (ceil(N/2)=513)
  float2 e = cis(-2.f * PI_F * (float)sh / 1025.f);
  return make_float2(1.f - e.x, -e.y);
}
__device__ float2 rot_sample(float th, int u, int v, int isY) {
  float c = __cosf(th), s = __sinf(th);
  const float ctr = 512.f;
  float fy =  c * ((float)u - ctr) + s * ((float)v - ctr) + ctr;
  float fx = -s * ((float)u - ctr) + c * ((float)v - ctr) + ctr;
  fy = fminf(fmaxf(fy, 0.f), 1024.f);
  fx = fminf(fmaxf(fx, 0.f), 1024.f);
  int y0 = (int)floorf(fy), x0 = (int)floorf(fx);
  int y1 = min(y0 + 1, 1024), x1 = min(x0 + 1, 1024);
  float wy = fy - (float)y0, wx = fx - (float)x0;
  float2 p00 = g_at(y0, x0, isY), p01 = g_at(y0, x1, isY);
  float2 p10 = g_at(y1, x0, isY), p11 = g_at(y1, x1, isY);
  float w00 = (1.f - wy) * (1.f - wx), w01 = (1.f - wy) * wx;
  float w10 = wy * (1.f - wx), w11 = wy * wx;
  return make_float2(w00 * p00.x + w01 * p01.x + w10 * p10.x + w11 * p11.x,
                     w00 * p00.y + w01 * p01.y + w10 * p10.y + w11 * p11.y);
}

// ================= Kernel 1: front complex MLP (WMMA) =================
__global__ __launch_bounds__(128) void k_front(
    const float* __restrict__ Xr, const float* __restrict__ Xi,
    const float2* __restrict__ fw0, const float2* __restrict__ fb0,
    const float2* __restrict__ fw1, const float2* __restrict__ fb1,
    const float2* __restrict__ lw,  const float2* __restrict__ lb,
    float2* __restrict__ Xf) {
  __shared__ __align__(32) _Float16 sBW[1024];     // fw1 B-frags
  __shared__ __align__(32) _Float16 fin[4][512];   // per-wave A-frag buffer
  __shared__ float2 xbuf[4][16];
  int tid = threadIdx.x, w = tid >> 5, lane = tid & 31;
  pack_b(fw1, 16, sBW, tid, 128);
  int p0 = (blockIdx.x * 4 + w) * 16;
  int m = lane & 15;
  float2 x = make_float2(Xr[p0 + m], Xi[p0 + m]);
  if (lane < 16) xbuf[w][m] = x;
  int cb = (lane < 16) ? 0 : 8;
#pragma unroll
  for (int c = 0; c < 8; ++c) {
    int ch = cb + c;
    frag_store(fin[w], 16, m, ch, crelu2(cadd(cmul(x, fw0[ch]), fb0[ch])));
  }
  __syncthreads();
  v8f dr, di;
  cgemm_f(fin[w], sBW, 1, lane, dr, di);
  int n = lane & 15, mb = (lane < 16) ? 0 : 8;
#pragma unroll
  for (int j = 0; j < 8; ++j) {
    int mm = mb + j;
    float2 xm = xbuf[w][mm];
    float2 v = make_float2(dr[j], di[j]);
    v = cadd(v, fb1[n]);
    v = cadd(v, cadd(cmul(xm, lw[n]), lb[n]));
    Xf[(long long)(p0 + mm) * 16 + n] = crelu2(v);
  }
}

// ================= Kernel 2: GNN gather + hier scatter =================
__global__ void k_gather(const float2* __restrict__ Xf, const float2* __restrict__ gnn_w,
                         const int* __restrict__ NI, const int* __restrict__ hmask,
                         const int* __restrict__ hind, float2* __restrict__ Xtv) {
  long long g = (long long)blockIdx.x * blockDim.x + threadIdx.x;
  if (g >= (long long)HW2 * 16) return;
  int j = (int)(g >> 4), ch = (int)(g & 15);
  int idx = hind[j];
  float2 acc;
  if (idx < N_G) {
    acc = make_float2(0.f, 0.f);
#pragma unroll
    for (int k = 0; k < 8; ++k) {
      int src = NI[(k + 1) * N_G + idx];
      acc = cadd(acc, cmul(Xf[(long long)src * 16 + ch], gnn_w[k * N_G + idx]));
    }
  } else {
    acc = Xf[(long long)hmask[idx - N_G] * 16 + ch];
  }
  Xtv[g] = acc;
}

// ================= Kernel 3: mega mlp2 chain + merge (WMMA) =================
// 64 threads = 2 waves, 16 px per wave.
__global__ __launch_bounds__(64) void k_mega(
    const float2* __restrict__ Xf, const float2* __restrict__ Xtv,
    const float2* __restrict__ ew1, const float2* __restrict__ eb1,
    const float2* __restrict__ ew2, const float2* __restrict__ eb2,
    const float2* __restrict__ lw1, const float2* __restrict__ lb1,
    const float2* __restrict__ lw2, const float2* __restrict__ lb2,
    const float2* __restrict__ mw1, const float2* __restrict__ mb1,
    const float2* __restrict__ mw2, const float2* __restrict__ mb2,
    const float2* __restrict__ mw3, const float2* __restrict__ mb3,
    float2* __restrict__ Xfr) {
  __shared__ __align__(32) _Float16 sBF[12 * 1024];  // ew1,ew2,lw1,lw2 B-frags
  __shared__ __align__(32) _Float16 sBM1[3 * 1024];  // mw1 (ldk=48) B-frags
  __shared__ __align__(32) _Float16 sBM2[1024];      // mw2 B-frags
  __shared__ __align__(32) _Float16 fin[2][512];     // A-frag staging
  __shared__ __align__(32) _Float16 fmid[2][512];
  __shared__ __align__(32) _Float16 ft[2][512];      // t accumulator
  __shared__ __align__(32) _Float16 fz[2][1536];     // zc (ldc=48, 3 chunks)
  __shared__ float2 sB[12][16];                      // eb1, eb2, lb1, lb2
  __shared__ float2 sMb1[16], sMb2[16], sMw3v[16];
  __shared__ float2 stvx[2][3][16], stvy[2][3][16];
  __shared__ float  seig[2][3][16];

  int tid = threadIdx.x, w = tid >> 5, lane = tid & 31;
  for (int a = 0; a < 3; ++a) {
    pack_b(ew1 + a * 256, 16, sBF + a * 1024,       tid, 64);
    pack_b(ew2 + a * 256, 16, sBF + (3 + a) * 1024, tid, 64);
    pack_b(lw1 + a * 256, 16, sBF + (6 + a) * 1024, tid, 64);
    pack_b(lw2 + a * 256, 16, sBF + (9 + a) * 1024, tid, 64);
  }
  pack_b(mw1, 48, sBM1, tid, 64);
  pack_b(mw2, 16, sBM2, tid, 64);
  if (tid < 48) {
    int a = tid >> 4, c2 = tid & 15;
    sB[a][c2] = eb1[tid];     sB[3 + a][c2] = eb2[tid];
    sB[6 + a][c2] = lb1[tid]; sB[9 + a][c2] = lb2[tid];
  }
  if (tid < 16) { sMb1[tid] = mb1[tid]; sMb2[tid] = mb2[tid]; sMw3v[tid] = mw3[tid]; }

  int p0 = (blockIdx.x * 2 + w) * 16;
  if (lane < 16) {
    int p = p0 + lane;
    int u = p / N_L, v = p % N_L;
#pragma unroll
    for (int a = 0; a < 3; ++a) {
      float A = kAngles[a] * (PI_F / 180.f);
      float th = atanf(3.f * tanf(A));
      float2 tx = rot_sample(th, u, v, 0);
      float2 ty = rot_sample(th, u, v, 1);
      stvx[w][a][lane] = tx; stvy[w][a][lane] = ty;
      seig[w][a][lane] = tx.x * tx.x + tx.y * tx.y + ty.x * ty.x + ty.y * ty.y;
    }
  }
  __syncthreads();

  int m = lane & 15, cb = (lane < 16) ? 0 : 8;
  int n = lane & 15, mb = (lane < 16) ? 0 : 8;
  v8f dr, di;

  for (int a = 0; a < 3; ++a) {
    // ---- x branch: x = Xf_tvx * TVFFTX ----
#pragma unroll
    for (int c = 0; c < 8; ++c) {
      int ch = cb + c;
      frag_store(fin[w], 16, m, ch, cmul(Xtv[(long long)(p0 + m) * 16 + ch], stvx[w][a][m]));
    }
    __syncthreads();
    cgemm_f(fin[w], sBF + a * 1024, 1, lane, dr, di);
#pragma unroll
    for (int j = 0; j < 8; ++j)
      frag_store(fmid[w], 16, mb + j, n,
                 crelu2(cadd(make_float2(dr[j], di[j]), sB[a][n])));
    __syncthreads();
    cgemm_f(fmid[w], sBF + (3 + a) * 1024, 1, lane, dr, di);
#pragma unroll
    for (int j = 0; j < 8; ++j) {
      int mm = mb + j;
      float2 vv = cadd(make_float2(dr[j], di[j]), sB[3 + a][n]);
      vv = crelu2(cmul(vv, conj2(stvx[w][a][mm])));
      frag_store(ft[w], 16, mm, n, vv);
    }
    __syncthreads();
    // ---- y branch: y = Xf * TVFFTY ----
#pragma unroll
    for (int c = 0; c < 8; ++c) {
      int ch = cb + c;
      frag_store(fin[w], 16, m, ch, cmul(Xf[(long long)(p0 + m) * 16 + ch], stvy[w][a][m]));
    }
    __syncthreads();
    cgemm_f(fin[w], sBF + a * 1024, 1, lane, dr, di);
#pragma unroll
    for (int j = 0; j < 8; ++j)
      frag_store(fmid[w], 16, mb + j, n,
                 crelu2(cadd(make_float2(dr[j], di[j]), sB[a][n])));
    __syncthreads();
    cgemm_f(fmid[w], sBF + (3 + a) * 1024, 1, lane, dr, di);
#pragma unroll
    for (int j = 0; j < 8; ++j) {
      int mm = mb + j;
      float2 vv = cadd(make_float2(dr[j], di[j]), sB[3 + a][n]);
      vv = crelu2(cmul(vv, conj2(stvy[w][a][mm])));
      frag_store(ft[w], 16, mm, n, cadd(frag_read(ft[w], 16, mm, n), vv));
    }
    __syncthreads();
    // ---- lw chain: z = crelu(mlp2(t)/eig) ----
    cgemm_f(ft[w], sBF + (6 + a) * 1024, 1, lane, dr, di);
#pragma unroll
    for (int j = 0; j < 8; ++j)
      frag_store(fmid[w], 16, mb + j, n,
                 crelu2(cadd(make_float2(dr[j], di[j]), sB[6 + a][n])));
    __syncthreads();
    cgemm_f(fmid[w], sBF + (9 + a) * 1024, 1, lane, dr, di);
#pragma unroll
    for (int j = 0; j < 8; ++j) {
      int mm = mb + j;
      float2 vv = cadd(make_float2(dr[j], di[j]), sB[9 + a][n]);
      float e = seig[w][a][mm];
      vv = crelu2(make_float2(vv.x / e, vv.y / e));
      frag_store(fz[w], 48, mm, a * 16 + n, vv);
    }
    __syncthreads();
  }
  // ---- merge: zc(48) -> mw1 -> mw2 -> mw3 ----
  cgemm_f(fz[w], sBM1, 3, lane, dr, di);
#pragma unroll
  for (int j = 0; j < 8; ++j)
    frag_store(fin[w], 16, mb + j, n,
               crelu2(cadd(make_float2(dr[j], di[j]), sMb1[n])));
  __syncthreads();
  cgemm_f(fin[w], sBM2, 1, lane, dr, di);
#pragma unroll
  for (int j = 0; j < 8; ++j)
    frag_store(fmid[w], 16, mb + j, n,
               crelu2(cadd(make_float2(dr[j], di[j]), sMb2[n])));
  __syncthreads();
  if (lane < 16) {
    float2 acc = mb3[0];
#pragma unroll
    for (int ch = 0; ch < 16; ++ch)
      acc = cadd(acc, cmul(frag_read(fmid[w], 16, lane, ch), sMw3v[ch]));
    Xfr[p0 + lane] = acc;
  }
}

// ================= Kernel 4: assemble Hermitian full spectrum =================
__global__ void k_assemble(const float2* __restrict__ Xfr, const float* __restrict__ ar,
                           const float* __restrict__ ai, float2* __restrict__ spec) {
  int i = blockIdx.x * blockDim.x + threadIdx.x;
  if (i >= NPIX) return;
  float2 v;
  if (i < HW2)       v = Xfr[i];
  else if (i == HW2) v = make_float2(ar[0], ai[0]);
  else { float2 t = Xfr[NPIX - 1 - i]; v = make_float2(t.x, -t.y); }
  spec[i] = v;
}

// ============ Kernel 5: iDFT pass as f32 WMMA GEMM (length 1025) ============
// OUT[t,n] = (1/1025) * sum_{k} src[t*1025 + (k+512)%1025] * e^{+2pi i k n/1025}
// Complex packed interleaved along K (re,im): K = 2052 (pad), chunks of 4.
// A frag (16x4 f32): lane m=l&15, slots = K {0,1} (l<16) / {2,3} (l>=16)
//   -> per lane one float2 (re,im) of complex index kc = 2*c + (l<16?0:1).
// B frag (4x16 f32): lane n=l&15, slots K {0,1}/{2,3} -> twiddle rotator
//   R = e^{2pi i kc n/1025}: real-out {R.x,-R.y}, imag-out {R.y,R.x}.
// mode 0: dst[n*1025 + t] = OUT (pass 1, transposed store)
// mode 1: outG[n*1025 + t] = |OUT| + alpha (pass 2)
__global__ __launch_bounds__(128) void k_idft_wmma(const float2* __restrict__ src,
                                                   float2* __restrict__ dst,
                                                   float* __restrict__ outG,
                                                   const float* __restrict__ alpha,
                                                   int mode) {
  int tid = threadIdx.x, w = tid >> 5, lane = tid & 31;
  int mtile = blockIdx.x;                 // 0..64 source-row tile
  int ntile = blockIdx.y * 4 + w;         // output-frequency tile
  if (ntile >= 65) return;
  int m = lane & 15;
  int row = min(mtile * 16 + m, 1024);
  const float2* rowp = src + (long long)row * 1025;
  int kcoff = (lane < 16) ? 0 : 1;
  int n = ntile * 16 + (lane & 15);
  float ang1 = 2.f * PI_F * (float)n / 1025.f;
  float2 R = cis(ang1 * (float)kcoff);     // e^{2pi i * kc * n / 1025}
  float2 step = cis(ang1 * 2.f);           // advance kc += 2 per chunk
  v8f dr = vzero8(), di = vzero8();
  int kc = kcoff;
  int col = (kc + 512) % 1025;
  for (int c = 0; c < 513; ++c) {
    float2 s = (kc <= 1024) ? rowp[col] : make_float2(0.f, 0.f);
    float2 Rc = (kc <= 1024) ? R : make_float2(0.f, 0.f);
    v2f a;  a[0] = s.x;   a[1] = s.y;
    v2f br; br[0] = Rc.x; br[1] = -Rc.y;
    v2f bi; bi[0] = Rc.y; bi[1] = Rc.x;
    dr = __builtin_amdgcn_wmma_f32_16x16x4_f32(false, a, false, br, (short)0, dr, false, false);
    di = __builtin_amdgcn_wmma_f32_16x16x4_f32(false, a, false, bi, (short)0, di, false, false);
    R = cmul(R, step);
    kc += 2;
    col += 2; if (col >= 1025) col -= 1025;
  }
  const float scale = 1.f / 1025.f;
  int mbase = (lane < 16) ? 0 : 8;
  if (n <= 1024) {
#pragma unroll
    for (int j = 0; j < 8; ++j) {
      int rr = mtile * 16 + mbase + j;
      if (rr <= 1024) {
        if (mode == 0) {
          dst[(long long)n * 1025 + rr] = make_float2(dr[j] * scale, di[j] * scale);
        } else {
          float re = dr[j] * scale, im = di[j] * scale;
          outG[(long long)n * 1025 + rr] = sqrtf(re * re + im * im) + alpha[0];
        }
      }
    }
  }
}

// ================= Kernel 7: guided-filter helpers =================
__global__ void k_copy(const float* __restrict__ s, float* __restrict__ d, int nelem) {
  int i = blockIdx.x * blockDim.x + threadIdx.x;
  if (i < nelem) d[i] = s[i];
}
__global__ void k_box(const float* __restrict__ src, const float* __restrict__ sub,
                      float* __restrict__ dst, float tanA) {
  int i = blockIdx.x * blockDim.x + threadIdx.x;
  if (i >= NPIX) return;
  int y = i / N_L, x = i % N_L;
  float acc = 0.f; int cnt = 0;
#pragma unroll 1
  for (int t = 0; t < 49; ++t) {
    int dy = t - 24;
    int dx = (int)lroundf((float)dy * tanA);
    int yy = y + dy, xx = x + dx;
    if (yy >= 0 && yy < M_L && xx >= 0 && xx < N_L) {
      float v = src[yy * N_L + xx];
      if (sub) v -= sub[yy * N_L + xx];
      acc += v; ++cnt;
    }
  }
  dst[i] = acc / (float)cnt;
}
__global__ void k_axpy(float* __restrict__ X, const float* __restrict__ b, int nelem) {
  int i = blockIdx.x * blockDim.x + threadIdx.x;
  if (i < nelem) X[i] += b[i];
}

// ================= Kernel 8: reflect grid-sample to 3072^2 =================
__device__ __forceinline__ float reflectf(float v, float span) {
  v = fabsf(v);
  v = fmodf(v, 2.f * span);
  return (v > span) ? (2.f * span - v) : v;
}
__global__ void k_gridsample(const float* __restrict__ X, const float* __restrict__ target,
                             const float* __restrict__ coor, const float* __restrict__ thr,
                             float* __restrict__ out) {
  int i = blockIdx.x * blockDim.x + threadIdx.x;
  if (i >= MH * NH) return;
  float gx = coor[2 * i + 0], gy = coor[2 * i + 1];
  float xs = reflectf((gx + 1.f) * 0.5f * 1024.f, 1024.f);
  float ys = reflectf((gy + 1.f) * 0.5f * 1024.f, 1024.f);
  float x0 = fminf(fmaxf(floorf(xs), 0.f), 1024.f);
  float y0 = fminf(fmaxf(floorf(ys), 0.f), 1024.f);
  int x0i = (int)x0, y0i = (int)y0;
  int x1i = min(x0i + 1, 1024), y1i = min(y0i + 1, 1024);
  float wx = xs - x0, wy = ys - y0;
  int i00 = y0i * N_L + x0i, i01 = y0i * N_L + x1i;
  int i10 = y1i * N_L + x0i, i11 = y1i * N_L + x1i;
  float v = (1.f - wy) * (1.f - wx) * (X[i00] - target[i00])
          + (1.f - wy) * wx        * (X[i01] - target[i01])
          + wy * (1.f - wx)        * (X[i10] - target[i10])
          + wy * wx                * (X[i11] - target[i11]);
  out[i] = v + thr[i];
}

// ================= host driver =================
extern "C" void kernel_launch(void* const* d_in, const int* in_sizes, int n_in,
                              void* d_out, int out_size, void* d_ws, size_t ws_size,
                              hipStream_t stream) {
  (void)in_sizes; (void)n_in; (void)out_size; (void)ws_size;
  const float*  Xr        = (const float*)d_in[0];
  const float*  Xi        = (const float*)d_in[1];
  const float*  aver_r    = (const float*)d_in[2];
  const float*  aver_i    = (const float*)d_in[3];
  const float*  target    = (const float*)d_in[4];
  const float*  target_hr = (const float*)d_in[5];
  const float*  coor      = (const float*)d_in[6];
  const float*  alpha     = (const float*)d_in[7];
  const float2* p_fw0 = (const float2*)d_in[8];
  const float2* p_fb0 = (const float2*)d_in[9];
  const float2* p_fw1 = (const float2*)d_in[10];
  const float2* p_fb1 = (const float2*)d_in[11];
  const float2* p_lw  = (const float2*)d_in[12];
  const float2* p_lb  = (const float2*)d_in[13];
  const float2* gnn_w = (const float2*)d_in[14];
  const float2* ew1 = (const float2*)d_in[15];
  const float2* eb1 = (const float2*)d_in[16];
  const float2* ew2 = (const float2*)d_in[17];
  const float2* eb2 = (const float2*)d_in[18];
  const float2* lw1 = (const float2*)d_in[19];
  const float2* lb1 = (const float2*)d_in[20];
  const float2* lw2 = (const float2*)d_in[21];
  const float2* lb2 = (const float2*)d_in[22];
  const float2* mw1 = (const float2*)d_in[23];
  const float2* mb1 = (const float2*)d_in[24];
  const float2* mw2 = (const float2*)d_in[25];
  const float2* mb2 = (const float2*)d_in[26];
  const float2* mw3 = (const float2*)d_in[27];
  const float2* mb3 = (const float2*)d_in[28];
  const int* NI    = (const int*)d_in[29];
  const int* hmask = (const int*)d_in[30];
  const int* hind  = (const int*)d_in[31];
  float* out = (float*)d_out;           // [NPIX outputGNNraw | MH*NH outputLR]

  char* wsp = (char*)d_ws;
  auto alloc = [&](size_t bytes) {
    char* p = wsp;
    wsp += (bytes + 255) & ~(size_t)255;
    return p;
  };
  float2* Xf   = (float2*)alloc((size_t)HW2 * 16 * sizeof(float2));
  float2* Xtv  = (float2*)alloc((size_t)HW2 * 16 * sizeof(float2));
  float2* Xfr  = (float2*)alloc((size_t)HW2 * sizeof(float2));
  float2* spec = (float2*)alloc((size_t)NPIX * sizeof(float2));
  float2* T1   = (float2*)alloc((size_t)NPIX * sizeof(float2));
  float*  Ximg = (float*)alloc((size_t)NPIX * sizeof(float));
  float*  b1   = (float*)alloc((size_t)NPIX * sizeof(float));
  float*  b2   = (float*)alloc((size_t)NPIX * sizeof(float));

  k_front<<<HW2 / 64, 128, 0, stream>>>(Xr, Xi, p_fw0, p_fb0, p_fw1, p_fb1,
                                        p_lw, p_lb, Xf);
  {
    long long tot = (long long)HW2 * 16;
    k_gather<<<(unsigned)((tot + 255) / 256), 256, 0, stream>>>(Xf, gnn_w, NI, hmask, hind, Xtv);
  }
  k_mega<<<HW2 / 32, 64, 0, stream>>>(Xf, Xtv, ew1, eb1, ew2, eb2,
                                      lw1, lb1, lw2, lb2,
                                      mw1, mb1, mw2, mb2, mw3, mb3, Xfr);
  k_assemble<<<(NPIX + 255) / 256, 256, 0, stream>>>(Xfr, aver_r, aver_i, spec);
  k_idft_wmma<<<dim3(65, 17), 128, 0, stream>>>(spec, T1, nullptr, nullptr, 0);
  k_idft_wmma<<<dim3(65, 17), 128, 0, stream>>>(T1, nullptr, out, alpha, 1);

  // guided filter: X = target; 3 angles x (2 boxes + add)
  k_copy<<<(NPIX + 255) / 256, 256, 0, stream>>>(target, Ximg, NPIX);
  static const float hAngles[3] = {-10.0f, 0.0f, 10.0f};
  for (int a = 0; a < 3; ++a) {
    float tA = 3.0f * tanf(hAngles[a] * (float)M_PI / 180.f);  // tan(atan(3 tanA))
    k_box<<<(NPIX + 255) / 256, 256, 0, stream>>>(out, Ximg, b1, tA);
    k_box<<<(NPIX + 255) / 256, 256, 0, stream>>>(b1, nullptr, b2, tA);
    k_axpy<<<(NPIX + 255) / 256, 256, 0, stream>>>(Ximg, b2, NPIX);
  }
  k_gridsample<<<(MH * NH + 255) / 256, 256, 0, stream>>>(Ximg, target, coor, target_hr,
                                                          out + NPIX);   // outputLR
}
